// HopfieldLayer_18769007083935
// MI455X (gfx1250) — matrix operand order repaired and checked by reading, not compiled, not run on hardware
//
#include <hip/hip_runtime.h>
#include <math.h>

// ---------------------------------------------------------------------------
// Fused Hopfield retrieval for MI455X (gfx1250, wave32, WMMA).
//   scores = beta * R @ Wk^T ; probs = softmax(scores) ; out = probs @ Wv
// v5: revert the 128-VGPR A-fragment hoist (it pushed allocation past 256
//     VGPRs and flooded the hot loop with s_set_vgpr_msb); keep the cheap
//     GEMM2 P-fragment hoist. Async double-buffered LDS fill retained.
// ---------------------------------------------------------------------------

#define BETA   0.1f
#define NDIM   512
#define WWIDTH 65536
#define BATCH  4096

#define BM       32     // batch rows per workgroup
#define WC       64     // memory slots per streamed chunk
#define RSTRIDE  520    // LDS row stride (bf16) for R / Wk rows (512 + 8 pad)
#define TSTRIDE  72     // LDS row stride (bf16) for Wv^T / P rows (64 + 8 pad)
#define LOG2E    1.44269504088896340736f

typedef __attribute__((ext_vector_type(16))) __bf16 v16bf;
typedef __attribute__((ext_vector_type(8)))  __bf16 v8bf;
typedef __attribute__((ext_vector_type(4)))  __bf16 v4bf;
typedef __attribute__((ext_vector_type(8)))  float  v8f;
typedef int b128_t __attribute__((vector_size(16)));   // 16-byte packet type

#if defined(__HIP_DEVICE_COMPILE__) && __has_builtin(__builtin_amdgcn_global_load_async_to_lds_b128)
#define HAVE_ASYNC 1
#else
#define HAVE_ASYNC 0
#endif

#if HAVE_ASYNC
typedef __attribute__((address_space(1))) b128_t* gpkt_t;   // global int4*
typedef __attribute__((address_space(3))) b128_t* lpkt_t;   // LDS int4*
#endif

__device__ __forceinline__ v16bf cat8(v8bf lo, v8bf hi) {
  return __builtin_shufflevector(lo, hi, 0,1,2,3,4,5,6,7,8,9,10,11,12,13,14,15);
}

__device__ __forceinline__ v8f wmma_bf16(v16bf a, v16bf b, v8f c) {
  return __builtin_amdgcn_wmma_f32_16x16x32_bf16(false, a, false, b, (short)0, c,
                                                 false, false);
}

// A-fragment (16x32 bf16): lane l: M=l%16; e 0..7 -> K=k0+(l/16)*8+e; 8..15 -> +16
__device__ __forceinline__ v16bf load_afrag(const __bf16* row, int k0) {
  v8bf lo = *(const v8bf*)(row + k0);
  v8bf hi = *(const v8bf*)(row + k0 + 16);
  return cat8(lo, hi);
}
// B-fragment (32x16 bf16): lane l: N=l%16; e 0..7 -> K=k0+(l/16)*8+e; 8..15 -> +16
__device__ __forceinline__ v16bf load_bfrag(const __bf16* row, int k0) {
  v8bf lo = *(const v8bf*)(row + k0);
  v8bf hi = *(const v8bf*)(row + k0 + 16);
  return cat8(lo, hi);
}

// 16-byte global -> LDS copy: async DMA (ASYNCcnt) when available.
__device__ __forceinline__ void copy_b128(const __bf16* g, __bf16* l) {
#if HAVE_ASYNC
  __builtin_amdgcn_global_load_async_to_lds_b128(
      (gpkt_t)(void*)g, (lpkt_t)(void*)l, 0, 0);
#else
  *(b128_t*)l = *(const b128_t*)g;
#endif
}

__device__ __forceinline__ void wait_async_copies() {
#if HAVE_ASYNC
#if __has_builtin(__builtin_amdgcn_s_wait_asynccnt)
  __builtin_amdgcn_s_wait_asynccnt(0);
#else
  asm volatile("s_wait_asynccnt 0x0" ::: "memory");
#endif
#endif
}

// ===========================================================================
// Pre-pass kernels (workspace path)
// ===========================================================================

// elementwise f32 -> bf16 (n4 = element count / 4)
__global__ void convert_f32_bf16_kernel(const float* __restrict__ src,
                                        __bf16* __restrict__ dst, int n4) {
  int i = blockIdx.x * blockDim.x + threadIdx.x;
  const int stride = gridDim.x * blockDim.x;
  for (; i < n4; i += stride) {
    const float4 v = *(const float4*)(src + (size_t)i * 4);
    v4bf b;
    b[0] = (__bf16)v.x; b[1] = (__bf16)v.y; b[2] = (__bf16)v.z; b[3] = (__bf16)v.w;
    *(v4bf*)(dst + (size_t)i * 4) = b;
  }
}

// Wv[65536][512] f32 -> WvT[512][65536] bf16, 64x64 tiles through LDS
__global__ __launch_bounds__(256)
void transpose_wv_bf16_kernel(const float* __restrict__ Wv,
                              __bf16* __restrict__ WvT) {
  __shared__ __bf16 tile[64 * 72];
  const int w0 = blockIdx.x * 64;
  const int n0 = blockIdx.y * 64;
  const int tid = threadIdx.x;
  #pragma unroll
  for (int j = 0; j < 4; ++j) {
    int idx = tid + 256 * j;          // 1024 float4 = 64 rows x 16
    int wi = idx >> 4, c4 = idx & 15;
    const float4 v = *(const float4*)(Wv + (size_t)(w0 + wi) * NDIM + n0 + c4 * 4);
    v4bf b;
    b[0] = (__bf16)v.x; b[1] = (__bf16)v.y; b[2] = (__bf16)v.z; b[3] = (__bf16)v.w;
    *(v4bf*)(tile + wi * 72 + c4 * 4) = b;
  }
  __syncthreads();
  #pragma unroll
  for (int j = 0; j < 2; ++j) {
    int idx = tid + 256 * j;          // 512 packets: ni(64) x pw(8)
    int ni = idx >> 3, pw = idx & 7;
    v8bf o;
    #pragma unroll
    for (int e = 0; e < 8; ++e) o[e] = tile[(pw * 8 + e) * 72 + ni];
    *(v8bf*)(WvT + (size_t)(n0 + ni) * WWIDTH + w0 + pw * 8) = o;
  }
}

// ===========================================================================
// Main fused kernel (bf16 workspace path, async double-buffered LDS fill)
// ===========================================================================

__device__ __forceinline__ void fill_chunk_bf16(const __bf16* __restrict__ Wkb,
                                                const __bf16* __restrict__ WvTg,
                                                __bf16* WkL, __bf16* WvL,
                                                int w0, int tid) {
  // Wk chunk: 64 rows x 512 bf16 = 4096 16B packets
  #pragma unroll
  for (int j = 0; j < 16; ++j) {
    int idx = tid + 256 * j;
    int s = idx >> 6, p = idx & 63;
    copy_b128(Wkb + (size_t)(w0 + s) * NDIM + p * 8, WkL + s * RSTRIDE + p * 8);
  }
  // WvT chunk: 512 rows x 64 bf16 = 4096 16B packets (contiguous per row!)
  #pragma unroll
  for (int j = 0; j < 16; ++j) {
    int idx = tid + 256 * j;
    int n = idx >> 3, p = idx & 7;
    copy_b128(WvTg + (size_t)n * WWIDTH + w0 + p * 8, WvL + n * TSTRIDE + p * 8);
  }
}

__global__ __launch_bounds__(256, 1)
void hopfield_main_bf16(const __bf16* __restrict__ Rb,
                        const __bf16* __restrict__ Wkb,
                        const __bf16* __restrict__ WvTg,
                        float* __restrict__ Out)
{
  extern __shared__ char smem_raw[];
  __bf16* R_lds = (__bf16*)smem_raw;                 // 32 x 520
  __bf16* WkB0  = R_lds + BM * RSTRIDE;              // 64 x 520 (buffer 0)
  __bf16* WkB1  = WkB0 + WC * RSTRIDE;               // 64 x 520 (buffer 1)
  __bf16* WvB0  = WkB1 + WC * RSTRIDE;               // 512 x 72 (buffer 0)
  __bf16* WvB1  = WvB0 + NDIM * TSTRIDE;             // 512 x 72 (buffer 1)
  __bf16* P_lds = WvB1 + NDIM * TSTRIDE;             // 32 x 72
  float*  redmax = (float*)(P_lds + BM * TSTRIDE);   // 32 x 4
  float*  redsum = redmax + BM * 4;                  // 32 x 4

  const int tid   = threadIdx.x;
  const int wave  = tid >> 5;
  const int lane  = tid & 31;
  const int rg    = wave >> 2;
  const int cg    = wave & 3;
  const int lhalf = lane >> 4;
  const int l16   = lane & 15;
  const int rowbase = rg * 16;
  const int wgrow   = blockIdx.x * BM;

  // R tile: 32 rows x 512 bf16 = 2048 16B packets
  #pragma unroll
  for (int j = 0; j < 8; ++j) {
    int idx = tid + 256 * j;
    int r = idx >> 6, p = idx & 63;
    copy_b128(Rb + (size_t)(wgrow + r) * NDIM + p * 8, R_lds + r * RSTRIDE + p * 8);
  }
  // first chunk into buffer 0
  fill_chunk_bf16(Wkb, WvTg, WkB0, WvB0, 0, tid);
  wait_async_copies();
  __syncthreads();

  v8f O[8];
  #pragma unroll
  for (int t = 0; t < 8; ++t)
    #pragma unroll
    for (int i = 0; i < 8; ++i) O[t][i] = 0.0f;
  float m_i[8], l_i[8];
  #pragma unroll
  for (int i = 0; i < 8; ++i) { m_i[i] = -__builtin_inff(); l_i[i] = 0.0f; }

  const __bf16* arow1 = R_lds + (rowbase + l16) * RSTRIDE + lhalf * 8;
  const int NCHUNK = WWIDTH / WC;

  for (int ci = 0; ci < NCHUNK; ++ci) {
    __bf16* WkCur = (ci & 1) ? WkB1 : WkB0;
    __bf16* WvCur = (ci & 1) ? WvB1 : WvB0;
    // prefetch next chunk into the other buffer (overlaps with WMMA below)
    if (ci + 1 < NCHUNK) {
      __bf16* WkNxt = (ci & 1) ? WkB0 : WkB1;
      __bf16* WvNxt = (ci & 1) ? WvB0 : WvB1;
      fill_chunk_bf16(Wkb, WvTg, WkNxt, WvNxt, (ci + 1) * WC, tid);
    }

    // ---- GEMM1: this wave's 16x16 score tile (rows rg, slots cg) ----------
    v8f S;
    #pragma unroll
    for (int i = 0; i < 8; ++i) S[i] = 0.0f;
    {
      const __bf16* brow = WkCur + (cg * 16 + l16) * RSTRIDE + lhalf * 8;
      #pragma unroll
      for (int kk = 0; kk < 16; ++kk) {
        const int k0 = kk * 32;
        S = wmma_bf16(load_afrag(arow1, k0), load_bfrag(brow, k0), S);
      }
    }

    // ---- online softmax (row stats span the 4 cg waves via LDS) -----------
    float sc[8];
    #pragma unroll
    for (int i = 0; i < 8; ++i) {
      float t = S[i] * BETA;
      float rm = t;
      rm = fmaxf(rm, __shfl_xor(rm, 1));
      rm = fmaxf(rm, __shfl_xor(rm, 2));
      rm = fmaxf(rm, __shfl_xor(rm, 4));
      rm = fmaxf(rm, __shfl_xor(rm, 8));
      if (l16 == 0) redmax[(rowbase + lhalf * 8 + i) * 4 + cg] = rm;
      S[i] = t;
    }
    __syncthreads();

    #pragma unroll
    for (int i = 0; i < 8; ++i) {
      const int row = rowbase + lhalf * 8 + i;
      float mc = fmaxf(fmaxf(redmax[row * 4 + 0], redmax[row * 4 + 1]),
                       fmaxf(redmax[row * 4 + 2], redmax[row * 4 + 3]));
      float mn = fmaxf(m_i[i], mc);
      sc[i] = __builtin_amdgcn_exp2f((m_i[i] - mn) * LOG2E);
      m_i[i] = mn;
      float pv = __builtin_amdgcn_exp2f((S[i] - mn) * LOG2E);
      float rs = pv;
      rs += __shfl_xor(rs, 1);
      rs += __shfl_xor(rs, 2);
      rs += __shfl_xor(rs, 4);
      rs += __shfl_xor(rs, 8);
      if (l16 == 0) redsum[row * 4 + cg] = rs;
      P_lds[row * TSTRIDE + cg * 16 + l16] = (__bf16)pv;
    }
    __syncthreads();

    #pragma unroll
    for (int i = 0; i < 8; ++i) {
      const int row = rowbase + lhalf * 8 + i;
      float rsum = (redsum[row * 4 + 0] + redsum[row * 4 + 1]) +
                   (redsum[row * 4 + 2] + redsum[row * 4 + 3]);
      l_i[i] = l_i[i] * sc[i] + rsum;
    }
    #pragma unroll
    for (int t = 0; t < 8; ++t)
      #pragma unroll
      for (int i = 0; i < 8; ++i) O[t][i] *= sc[i];

    // ---- GEMM2: O[16x128] += P[16x64] @ Wv_chunk[64x128] ------------------
    {
      const __bf16* arow2 = P_lds + (rowbase + l16) * TSTRIDE + lhalf * 8;
      const v16bf Pa0 = load_afrag(arow2, 0);    // shared across all 8 tiles
      const v16bf Pa1 = load_afrag(arow2, 32);
      #pragma unroll
      for (int t = 0; t < 8; ++t) {
        const __bf16* brow2 = WvCur + (cg * 128 + t * 16 + l16) * TSTRIDE + lhalf * 8;
        O[t] = wmma_bf16(Pa0, load_bfrag(brow2, 0), O[t]);
        O[t] = wmma_bf16(Pa1, load_bfrag(brow2, 32), O[t]);
      }
    }
    wait_async_copies();   // next chunk's DMA done
    __syncthreads();       // everyone done with cur buffers + P
  }

  #pragma unroll
  for (int t = 0; t < 8; ++t) {
    const int col = cg * 128 + t * 16 + l16;
    #pragma unroll
    for (int i = 0; i < 8; ++i) {
      const int row = wgrow + rowbase + lhalf * 8 + i;
      Out[(size_t)row * NDIM + col] = O[t][i] / l_i[i];
    }
  }
}

// ===========================================================================
// Fallback: fused conversion kernel (no workspace needed)
// ===========================================================================

__global__ __launch_bounds__(256, 1)
void hopfield_fused_fallback(const float* __restrict__ R,
                             const float* __restrict__ Wk,
                             const float* __restrict__ Wv,
                             float* __restrict__ Out)
{
  extern __shared__ char smem_raw[];
  __bf16* R_lds  = (__bf16*)smem_raw;
  __bf16* Wk_lds = R_lds  + BM   * RSTRIDE;
  __bf16* WvT    = Wk_lds + WC   * RSTRIDE;
  __bf16* P_lds  = WvT    + NDIM * TSTRIDE;
  float*  redmax = (float*)(P_lds + BM * TSTRIDE);
  float*  redsum = redmax + BM * 4;

  const int tid   = threadIdx.x;
  const int wave  = tid >> 5;
  const int lane  = tid & 31;
  const int rg    = wave >> 2;
  const int cg    = wave & 3;
  const int lhalf = lane >> 4;
  const int l16   = lane & 15;
  const int rowbase = rg * 16;
  const int wgrow   = blockIdx.x * BM;

  #pragma unroll
  for (int j = 0; j < 16; ++j) {
    int idx = tid + 256 * j;
    int r = idx >> 7, c4 = idx & 127;
    const float4 v = *(const float4*)(R + (size_t)(wgrow + r) * NDIM + c4 * 4);
    v4bf b;
    b[0] = (__bf16)v.x; b[1] = (__bf16)v.y; b[2] = (__bf16)v.z; b[3] = (__bf16)v.w;
    *(v4bf*)(R_lds + r * RSTRIDE + c4 * 4) = b;
  }

  v8f O[8];
  #pragma unroll
  for (int t = 0; t < 8; ++t)
    #pragma unroll
    for (int i = 0; i < 8; ++i) O[t][i] = 0.0f;
  float m_i[8], l_i[8];
  #pragma unroll
  for (int i = 0; i < 8; ++i) { m_i[i] = -__builtin_inff(); l_i[i] = 0.0f; }

  const __bf16* arow1 = R_lds + (rowbase + l16) * RSTRIDE + lhalf * 8;

  for (int w0 = 0; w0 < WWIDTH; w0 += WC) {
    #pragma unroll
    for (int j = 0; j < 32; ++j) {
      int idx = tid + 256 * j;
      int s = idx >> 7, c4 = idx & 127;
      const size_t gbase = (size_t)(w0 + s) * NDIM + c4 * 4;
      const float4 a = *(const float4*)(Wk + gbase);
      v4bf b;
      b[0] = (__bf16)a.x; b[1] = (__bf16)a.y; b[2] = (__bf16)a.z; b[3] = (__bf16)a.w;
      *(v4bf*)(Wk_lds + s * RSTRIDE + c4 * 4) = b;
      const float4 c = *(const float4*)(Wv + gbase);
      int n = c4 * 4;
      WvT[(n + 0) * TSTRIDE + s] = (__bf16)c.x;
      WvT[(n + 1) * TSTRIDE + s] = (__bf16)c.y;
      WvT[(n + 2) * TSTRIDE + s] = (__bf16)c.z;
      WvT[(n + 3) * TSTRIDE + s] = (__bf16)c.w;
    }
    __syncthreads();

    v8f S;
    #pragma unroll
    for (int i = 0; i < 8; ++i) S[i] = 0.0f;
    {
      const __bf16* brow = Wk_lds + (cg * 16 + l16) * RSTRIDE + lhalf * 8;
      #pragma unroll
      for (int kk = 0; kk < 16; ++kk)
        S = wmma_bf16(load_afrag(arow1, kk * 32), load_bfrag(brow, kk * 32), S);
    }

    float sc[8];
    #pragma unroll
    for (int i = 0; i < 8; ++i) {
      float t = S[i] * BETA;
      float rm = t;
      rm = fmaxf(rm, __shfl_xor(rm, 1));
      rm = fmaxf(rm, __shfl_xor(rm, 2));
      rm = fmaxf(rm, __shfl_xor(rm, 4));
      rm = fmaxf(rm, __shfl_xor(rm, 8));
      if (l16 == 0) redmax[(rowbase + lhalf * 8 + i) * 4 + cg] = rm;
      S[i] = t;
    }
    __syncthreads();
    #pragma unroll
    for (int i = 0; i < 8; ++i) {
      const int row = rowbase + lhalf * 8 + i;
      float mc = fmaxf(fmaxf(redmax[row * 4 + 0], redmax[row * 4 + 1]),
                       fmaxf(redmax[row * 4 + 2], redmax[row * 4 + 3]));
      float mn = fmaxf(m_i[i], mc);
      sc[i] = __builtin_amdgcn_exp2f((m_i[i] - mn) * LOG2E);
      m_i[i] = mn;
      float pv = __builtin_amdgcn_exp2f((S[i] - mn) * LOG2E);
      float rs = pv;
      rs += __shfl_xor(rs, 1);
      rs += __shfl_xor(rs, 2);
      rs += __shfl_xor(rs, 4);
      rs += __shfl_xor(rs, 8);
      if (l16 == 0) redsum[row * 4 + cg] = rs;
      P_lds[row * TSTRIDE + cg * 16 + l16] = (__bf16)pv;
    }
    __syncthreads();
    #pragma unroll
    for (int i = 0; i < 8; ++i) {
      const int row = rowbase + lhalf * 8 + i;
      float rsum = (redsum[row * 4 + 0] + redsum[row * 4 + 1]) +
                   (redsum[row * 4 + 2] + redsum[row * 4 + 3]);
      l_i[i] = l_i[i] * sc[i] + rsum;
    }
    #pragma unroll
    for (int t = 0; t < 8; ++t)
      #pragma unroll
      for (int i = 0; i < 8; ++i) O[t][i] *= sc[i];

    {
      const __bf16* arow2 = P_lds + (rowbase + l16) * TSTRIDE + lhalf * 8;
      const v16bf Pa0 = load_afrag(arow2, 0);
      const v16bf Pa1 = load_afrag(arow2, 32);
      #pragma unroll
      for (int t = 0; t < 8; ++t) {
        const __bf16* brow2 = WvT + (cg * 128 + t * 16 + l16) * TSTRIDE + lhalf * 8;
        O[t] = wmma_bf16(Pa0, load_bfrag(brow2, 0), O[t]);
        O[t] = wmma_bf16(Pa1, load_bfrag(brow2, 32), O[t]);
      }
    }
    __syncthreads();
  }

  #pragma unroll
  for (int t = 0; t < 8; ++t) {
    const int col = cg * 128 + t * 16 + l16;
    #pragma unroll
    for (int i = 0; i < 8; ++i) {
      const int row = wgrow + rowbase + lhalf * 8 + i;
      Out[(size_t)row * NDIM + col] = O[t][i] / l_i[i];
    }
  }
}

// ===========================================================================

extern "C" void kernel_launch(void* const* d_in, const int* in_sizes, int n_in,
                              void* d_out, int out_size, void* d_ws, size_t ws_size,
                              hipStream_t stream) {
  (void)in_sizes; (void)n_in; (void)out_size;
  const float* R  = (const float*)d_in[0];
  const float* Wk = (const float*)d_in[1];
  const float* Wv = (const float*)d_in[2];
  float* Out = (float*)d_out;

  const size_t nR  = (size_t)BATCH * NDIM;     //  2.1 M elems
  const size_t nW  = (size_t)WWIDTH * NDIM;    // 33.5 M elems
  const size_t wsNeed = (nR + 2 * nW) * sizeof(__bf16);   // ~132 MB

  if (ws_size >= wsNeed) {
    __bf16* Rb   = (__bf16*)d_ws;
    __bf16* Wkb  = Rb + nR;
    __bf16* WvTg = Wkb + nW;

    convert_f32_bf16_kernel<<<dim3(1024), dim3(256), 0, stream>>>(R, Rb, (int)(nR / 4));
    convert_f32_bf16_kernel<<<dim3(4096), dim3(256), 0, stream>>>(Wk, Wkb, (int)(nW / 4));
    transpose_wv_bf16_kernel<<<dim3(WWIDTH / 64, NDIM / 64), dim3(256), 0, stream>>>(Wv, WvTg);

    // LDS: R(32x520) + 2x Wk(64x520) + 2x WvT(512x72) + P(32x72) bf16 + red = 319488 B
    const size_t shmem = (size_t)(BM * RSTRIDE + 2 * WC * RSTRIDE +
                                  2 * NDIM * TSTRIDE + BM * TSTRIDE) * sizeof(__bf16) +
                         (size_t)(2 * BM * 4) * sizeof(float);
    (void)hipFuncSetAttribute((const void*)hopfield_main_bf16,
                              hipFuncAttributeMaxDynamicSharedMemorySize, (int)shmem);
    hopfield_main_bf16<<<dim3(BATCH / BM), dim3(256), shmem, stream>>>(Rb, Wkb, WvTg, Out);
  } else {
    const size_t shmem = (size_t)(BM * RSTRIDE + WC * RSTRIDE + NDIM * TSTRIDE +
                                  BM * TSTRIDE) * sizeof(__bf16) +
                         (size_t)(2 * BM * 4) * sizeof(float);
    (void)hipFuncSetAttribute((const void*)hopfield_fused_fallback,
                              hipFuncAttributeMaxDynamicSharedMemorySize, (int)shmem);
    hopfield_fused_fallback<<<dim3(BATCH / BM), dim3(256), shmem, stream>>>(R, Wk, Wv, Out);
  }
}
